// NodeBlock_82575041233373
// MI455X (gfx1250) — matrix operand order repaired and checked by reading, not compiled, not run on hardware
//
#include <hip/hip_runtime.h>
#include <hip/hip_bf16.h>

// Fused GNN NodeBlock for MI455X (gfx1250, wave32, WMMA f16->f32).
//
// Input layout assumption (setup_inputs dict order, params flattened
// depth-first in insertion order):
//   d_in[0]=pos [N,3] f32, d_in[1]=ang [N,4] f32, d_in[2]=edge_index [2,E] i32,
//   d_in[3]=edge_attr [E,64] f32,
//   d_in[4..11]  = pos1 {W1,b1,gamma,beta,W2,b2,W3,b3}
//   d_in[12..19] = pos2, d_in[20..27] = ang1, d_in[28..35] = ang2
// d_out = [u (N*3) | phi (N*4)] f32.

typedef _Float16 h16;
typedef __attribute__((ext_vector_type(16))) _Float16 v16h;
typedef __attribute__((ext_vector_type(8)))  float    v8f;

#define HD 128          // hidden dim
#define ED 64           // edge dim
#define BM 128          // rows (edges/nodes) per block tile
#define K1PAD 96        // layer-1 K padded (67/68 -> 96)
#define NTHR 256        // 8 wave32 waves per block
#define BN_EPS 1e-5f

// ---------------------------------------------------------------------------
// Fragment loaders (LDS -> VGPR) matching CDNA5 WMMA 16-bit layouts.
// A (16x32 f16): lanes 0-15 M=0..15 K-half 0, lanes 16-31 M=0..15 K-half 1.
//   VGPR v<4 -> K = 2v..2v+1 (+8*half), v>=4 -> K = 16+2(v-4).. (+8*half)
// B (32x16 f16): lane l holds N = l%16, K in the same order as A
//   (weights stored K-major in LDS: Wt[k][n]).
// ---------------------------------------------------------------------------
__device__ __forceinline__ v16h lds_load_a(const h16* __restrict__ X, int stride,
                                           int m, int k0, int g) {
  const h16* p = X + m * stride + k0 + 8 * g;
  v16h a;
#pragma unroll
  for (int v = 0; v < 4; ++v) { a[2 * v] = p[2 * v]; a[2 * v + 1] = p[2 * v + 1]; }
#pragma unroll
  for (int v = 0; v < 4; ++v) { a[8 + 2 * v] = p[16 + 2 * v]; a[9 + 2 * v] = p[17 + 2 * v]; }
  return a;
}

__device__ __forceinline__ v16h lds_load_b(const h16* __restrict__ Wt, int nstride,
                                           int k0, int n0, int lane) {
  const h16* p = Wt + (size_t)(k0 + 8 * (lane >> 4)) * nstride + n0 + (lane & 15);
  v16h b;
#pragma unroll
  for (int s = 0; s < 8; ++s) { b[s] = p[s * nstride]; b[8 + s] = p[(16 + s) * nstride]; }
  return b;
}

// One wave computes a 16-row strip (rows row0..row0+15) x NT*16 cols, K unrolled.
template <int K, int NT>
__device__ __forceinline__ void gemm_strip(const h16* __restrict__ X, int xs,
                                           const h16* __restrict__ Wt, int ws,
                                           int row0, int lane, v8f* acc) {
#pragma unroll
  for (int k0 = 0; k0 < K; k0 += 32) {
    v16h a = lds_load_a(X, xs, row0 + (lane & 15), k0, lane >> 4);
#pragma unroll
    for (int nt = 0; nt < NT; ++nt) {
      v16h b = lds_load_b(Wt, ws, k0, nt * 16, lane);
      acc[nt] = __builtin_amdgcn_wmma_f32_16x16x32_f16(
          false, a, false, b, (short)0, acc[nt], false, false);
    }
  }
}

// Global f32 row-major W[nout][k] -> LDS f16 K-major Wt[kpad][npad], zero padded.
__device__ __forceinline__ void load_wt(h16* __restrict__ Wt, const float* __restrict__ W,
                                        int nout, int npad, int k, int kpad, int tid) {
  for (int idx = tid; idx < kpad * npad; idx += NTHR) {
    int kk = idx / npad, nn = idx - kk * npad;
    float v = (kk < k && nn < nout) ? W[nn * k + kk] : 0.0f;
    Wt[idx] = (h16)v;
  }
}

// ---------------------------------------------------------------------------
// Edge pass 1: h1 = relu(X @ W1^T + b1), accumulate per-channel sum / sumsq.
// ---------------------------------------------------------------------------
template <int ND>
__global__ __launch_bounds__(NTHR)
void edge_pass1(const float* __restrict__ nodef, const float* __restrict__ eattr,
                const int* __restrict__ rowIdx,
                const float* __restrict__ W1, const float* __restrict__ b1,
                float* __restrict__ statSum, float* __restrict__ statSq, int E) {
  __shared__ __align__(32) h16 X[BM * K1PAD];
  __shared__ __align__(32) h16 W1t[K1PAD * HD];
  __shared__ __align__(32) h16 H[BM * HD];
  __shared__ float b1s[HD], redS[HD], redQ[HD];

  const int tid = threadIdx.x, lane = tid & 31, w = tid >> 5;
  const int e0 = blockIdx.x * BM;

  load_wt(W1t, W1, HD, HD, ND + ED, K1PAD, tid);
  if (tid < HD) b1s[tid] = b1[tid];
  if (tid == 0 && e0 + BM < E)  // warm next tile of edge_attr into caches
    __builtin_prefetch(eattr + (size_t)(e0 + BM) * ED, 0, 1);

  for (int idx = tid; idx < BM * K1PAD; idx += NTHR) {
    int m = idx / K1PAD, j = idx - m * K1PAD;
    int e = e0 + m;
    float v = 0.0f;
    if (e < E) {
      if (j < ND)            v = nodef[(size_t)rowIdx[e] * ND + j];
      else if (j < ND + ED)  v = eattr[(size_t)e * ED + (j - ND)];
    }
    X[idx] = (h16)v;
  }
  __syncthreads();

  v8f acc[8] = {};
  gemm_strip<K1PAD, 8>(X, K1PAD, W1t, HD, w * 16, lane, acc);

  const int g = lane >> 4, nl = lane & 15;
#pragma unroll
  for (int nt = 0; nt < 8; ++nt) {
    int n = nt * 16 + nl;
    float bb = b1s[n];
#pragma unroll
    for (int v = 0; v < 8; ++v) {
      int m = w * 16 + g * 8 + v;
      H[m * HD + n] = (h16)fmaxf(acc[nt][v] + bb, 0.0f);
    }
  }
  __syncthreads();

  int mlim = E - e0; if (mlim > BM) mlim = BM;
  int c = tid & (HD - 1), p = tid >> 7;
  float s = 0.0f, q = 0.0f;
  for (int m = p * (BM / 2); m < p * (BM / 2) + (BM / 2); ++m) {
    if (m < mlim) { float hv = (float)H[m * HD + c]; s += hv; q += hv * hv; }
  }
  if (p == 1) { redS[c] = s; redQ[c] = q; }
  __syncthreads();
  if (p == 0) {
    atomicAdd(&statSum[c], s + redS[c]);
    atomicAdd(&statSq[c],  q + redQ[c]);
  }
}

// ---------------------------------------------------------------------------
// BN finalize: s = gamma*rsqrt(var+eps), t = beta - mu*s   (128 channels)
// ---------------------------------------------------------------------------
__global__ void bn_finalize(const float* __restrict__ sum, const float* __restrict__ sq,
                            const float* __restrict__ gamma, const float* __restrict__ beta,
                            float count, float* __restrict__ sOut, float* __restrict__ tOut) {
  int c = threadIdx.x;
  if (c < HD) {
    float mu = sum[c] / count;
    float var = sq[c] / count - mu * mu;
    float s = gamma[c] * rsqrtf(var + BN_EPS);
    sOut[c] = s;
    tOut[c] = beta[c] - mu * s;
  }
}

// ---------------------------------------------------------------------------
// Edge pass 2: full 3-layer MLP (BN folded) + atomic scatter into accum[col].
// ---------------------------------------------------------------------------
template <int ND>
__global__ __launch_bounds__(NTHR)
void edge_pass2(const float* __restrict__ nodef, const float* __restrict__ eattr,
                const int* __restrict__ rowIdx, const int* __restrict__ colIdx,
                const float* __restrict__ W1, const float* __restrict__ b1,
                const float* __restrict__ bnS, const float* __restrict__ bnT,
                const float* __restrict__ W2, const float* __restrict__ b2,
                const float* __restrict__ W3, const float* __restrict__ b3,
                float* __restrict__ accum, int E) {
  // X + W1t region is dead after layer 1; reuse it for H2.
  __shared__ __align__(32) h16 XW[BM * K1PAD + K1PAD * HD];
  __shared__ __align__(32) h16 H[BM * HD];
  __shared__ __align__(32) h16 W2t[HD * HD];
  __shared__ __align__(32) h16 W3t[HD * ED];
  __shared__ float b1s[HD], ss[HD], ts[HD], b2s[HD], b3s[ED];

  h16* X   = XW;
  h16* W1t = XW + BM * K1PAD;
  h16* H2  = XW;

  const int tid = threadIdx.x, lane = tid & 31, w = tid >> 5;
  const int e0 = blockIdx.x * BM;
  const int g = lane >> 4, nl = lane & 15;

  load_wt(W1t, W1, HD, HD, ND + ED, K1PAD, tid);
  load_wt(W2t, W2, HD, HD, HD, HD, tid);
  load_wt(W3t, W3, ED, ED, HD, HD, tid);
  if (tid < HD) { b1s[tid] = b1[tid]; ss[tid] = bnS[tid]; ts[tid] = bnT[tid]; b2s[tid] = b2[tid]; }
  if (tid < ED) b3s[tid] = b3[tid];

  for (int idx = tid; idx < BM * K1PAD; idx += NTHR) {
    int m = idx / K1PAD, j = idx - m * K1PAD;
    int e = e0 + m;
    float v = 0.0f;
    if (e < E) {
      if (j < ND)            v = nodef[(size_t)rowIdx[e] * ND + j];
      else if (j < ND + ED)  v = eattr[(size_t)e * ED + (j - ND)];
    }
    X[idx] = (h16)v;
  }
  __syncthreads();

  { // layer 1 + folded BatchNorm
    v8f acc[8] = {};
    gemm_strip<K1PAD, 8>(X, K1PAD, W1t, HD, w * 16, lane, acc);
#pragma unroll
    for (int nt = 0; nt < 8; ++nt) {
      int n = nt * 16 + nl;
      float bb = b1s[n], sc = ss[n], sh = ts[n];
#pragma unroll
      for (int v = 0; v < 8; ++v) {
        int m = w * 16 + g * 8 + v;
        H[m * HD + n] = (h16)(fmaxf(acc[nt][v] + bb, 0.0f) * sc + sh);
      }
    }
  }
  __syncthreads();

  { // layer 2 (writes H2 which aliases the now-dead X/W1t region)
    v8f acc[8] = {};
    gemm_strip<HD, 8>(H, HD, W2t, HD, w * 16, lane, acc);
#pragma unroll
    for (int nt = 0; nt < 8; ++nt) {
      int n = nt * 16 + nl;
      float bb = b2s[n];
#pragma unroll
      for (int v = 0; v < 8; ++v) {
        int m = w * 16 + g * 8 + v;
        H2[m * HD + n] = (h16)fmaxf(acc[nt][v] + bb, 0.0f);
      }
    }
  }
  __syncthreads();

  { // layer 3 + scatter-add into node accumulators
    v8f acc[4] = {};
    gemm_strip<HD, 4>(H2, HD, W3t, ED, w * 16, lane, acc);
    int cols[8]; bool valid[8];
#pragma unroll
    for (int v = 0; v < 8; ++v) {
      int m = w * 16 + g * 8 + v;
      int e = e0 + m;
      valid[v] = (e < E);
      cols[v] = valid[v] ? colIdx[e] : 0;
    }
#pragma unroll
    for (int nt = 0; nt < 4; ++nt) {
      int n = nt * 16 + nl;
      float bb = b3s[n];
#pragma unroll
      for (int v = 0; v < 8; ++v) {
        if (valid[v]) atomicAdd(&accum[(size_t)cols[v] * ED + n], acc[nt][v] + bb);
      }
    }
  }
}

// ---------------------------------------------------------------------------
// Node pass 1: X = [nodef | accum/cnt], stats of relu(layer1).
// ---------------------------------------------------------------------------
template <int ND>
__global__ __launch_bounds__(NTHR)
void node_pass1(const float* __restrict__ nodef, const float* __restrict__ accum,
                const float* __restrict__ cnt,
                const float* __restrict__ W1, const float* __restrict__ b1,
                float* __restrict__ statSum, float* __restrict__ statSq, int N) {
  __shared__ __align__(32) h16 X[BM * K1PAD];
  __shared__ __align__(32) h16 W1t[K1PAD * HD];
  __shared__ __align__(32) h16 H[BM * HD];
  __shared__ float b1s[HD], redS[HD], redQ[HD], inv[BM];

  const int tid = threadIdx.x, lane = tid & 31, w = tid >> 5;
  const int i0 = blockIdx.x * BM;

  load_wt(W1t, W1, HD, HD, ND + ED, K1PAD, tid);
  if (tid < HD) b1s[tid] = b1[tid];
  if (tid < BM) {
    int i = i0 + tid;
    float c = (i < N) ? cnt[i] : 1.0f;
    inv[tid] = 1.0f / fmaxf(c, 1.0f);
  }
  __syncthreads();

  for (int idx = tid; idx < BM * K1PAD; idx += NTHR) {
    int m = idx / K1PAD, j = idx - m * K1PAD;
    int i = i0 + m;
    float v = 0.0f;
    if (i < N) {
      if (j < ND)            v = nodef[(size_t)i * ND + j];
      else if (j < ND + ED)  v = accum[(size_t)i * ED + (j - ND)] * inv[m];
    }
    X[idx] = (h16)v;
  }
  __syncthreads();

  v8f acc[8] = {};
  gemm_strip<K1PAD, 8>(X, K1PAD, W1t, HD, w * 16, lane, acc);

  const int g = lane >> 4, nl = lane & 15;
#pragma unroll
  for (int nt = 0; nt < 8; ++nt) {
    int n = nt * 16 + nl;
    float bb = b1s[n];
#pragma unroll
    for (int v = 0; v < 8; ++v) {
      int m = w * 16 + g * 8 + v;
      H[m * HD + n] = (h16)fmaxf(acc[nt][v] + bb, 0.0f);
    }
  }
  __syncthreads();

  int mlim = N - i0; if (mlim > BM) mlim = BM;
  int c = tid & (HD - 1), p = tid >> 7;
  float s = 0.0f, q = 0.0f;
  for (int m = p * (BM / 2); m < p * (BM / 2) + (BM / 2); ++m) {
    if (m < mlim) { float hv = (float)H[m * HD + c]; s += hv; q += hv * hv; }
  }
  if (p == 1) { redS[c] = s; redQ[c] = q; }
  __syncthreads();
  if (p == 0) {
    atomicAdd(&statSum[c], s + redS[c]);
    atomicAdd(&statSq[c],  q + redQ[c]);
  }
}

// ---------------------------------------------------------------------------
// Node pass 2: full MLP (BN folded) + residual, writes final output.
// ---------------------------------------------------------------------------
template <int ND>
__global__ __launch_bounds__(NTHR)
void node_pass2(const float* __restrict__ nodef, const float* __restrict__ accum,
                const float* __restrict__ cnt,
                const float* __restrict__ W1, const float* __restrict__ b1,
                const float* __restrict__ bnS, const float* __restrict__ bnT,
                const float* __restrict__ W2, const float* __restrict__ b2,
                const float* __restrict__ W3, const float* __restrict__ b3,
                float* __restrict__ out, int N) {
  __shared__ __align__(32) h16 XW[BM * K1PAD + K1PAD * HD];  // X+W1t, reused as H2
  __shared__ __align__(32) h16 H[BM * HD];
  __shared__ __align__(32) h16 W2t[HD * HD];
  __shared__ __align__(32) h16 W3t[HD * 16];
  __shared__ float b1s[HD], ss[HD], ts[HD], b2s[HD], b3s[16], inv[BM];

  h16* X   = XW;
  h16* W1t = XW + BM * K1PAD;
  h16* H2  = XW;

  const int tid = threadIdx.x, lane = tid & 31, w = tid >> 5;
  const int i0 = blockIdx.x * BM;
  const int g = lane >> 4, nl = lane & 15;

  load_wt(W1t, W1, HD, HD, ND + ED, K1PAD, tid);
  load_wt(W2t, W2, HD, HD, HD, HD, tid);
  load_wt(W3t, W3, ND, 16, HD, HD, tid);  // pad DOUT (3/4) -> 16 with zeros
  if (tid < HD) { b1s[tid] = b1[tid]; ss[tid] = bnS[tid]; ts[tid] = bnT[tid]; b2s[tid] = b2[tid]; }
  if (tid < 16) b3s[tid] = (tid < ND) ? b3[tid] : 0.0f;
  if (tid < BM) {
    int i = i0 + tid;
    float c = (i < N) ? cnt[i] : 1.0f;
    inv[tid] = 1.0f / fmaxf(c, 1.0f);
  }
  __syncthreads();

  for (int idx = tid; idx < BM * K1PAD; idx += NTHR) {
    int m = idx / K1PAD, j = idx - m * K1PAD;
    int i = i0 + m;
    float v = 0.0f;
    if (i < N) {
      if (j < ND)            v = nodef[(size_t)i * ND + j];
      else if (j < ND + ED)  v = accum[(size_t)i * ED + (j - ND)] * inv[m];
    }
    X[idx] = (h16)v;
  }
  __syncthreads();

  { // layer 1 + folded BN
    v8f acc[8] = {};
    gemm_strip<K1PAD, 8>(X, K1PAD, W1t, HD, w * 16, lane, acc);
#pragma unroll
    for (int nt = 0; nt < 8; ++nt) {
      int n = nt * 16 + nl;
      float bb = b1s[n], sc = ss[n], sh = ts[n];
#pragma unroll
      for (int v = 0; v < 8; ++v) {
        int m = w * 16 + g * 8 + v;
        H[m * HD + n] = (h16)(fmaxf(acc[nt][v] + bb, 0.0f) * sc + sh);
      }
    }
  }
  __syncthreads();

  { // layer 2
    v8f acc[8] = {};
    gemm_strip<HD, 8>(H, HD, W2t, HD, w * 16, lane, acc);
#pragma unroll
    for (int nt = 0; nt < 8; ++nt) {
      int n = nt * 16 + nl;
      float bb = b2s[n];
#pragma unroll
      for (int v = 0; v < 8; ++v) {
        int m = w * 16 + g * 8 + v;
        H2[m * HD + n] = (h16)fmaxf(acc[nt][v] + bb, 0.0f);
      }
    }
  }
  __syncthreads();

  { // layer 3 (single 16-col N tile, DOUT = ND) + residual
    v8f acc[1] = {};
    gemm_strip<HD, 1>(H2, HD, W3t, 16, w * 16, lane, acc);
    if (nl < ND) {
#pragma unroll
      for (int v = 0; v < 8; ++v) {
        int m = w * 16 + g * 8 + v;
        int i = i0 + m;
        if (i < N) {
          out[(size_t)i * ND + nl] =
              nodef[(size_t)i * ND + nl] + acc[0][v] + b3s[nl];
        }
      }
    }
  }
}

// ---------------------------------------------------------------------------
// In-degree histogram (for scatter-mean denominator).
// ---------------------------------------------------------------------------
__global__ void count_edges(const int* __restrict__ col, float* __restrict__ cnt, int E) {
  int e = blockIdx.x * blockDim.x + threadIdx.x;
  if (e < E) atomicAdd(&cnt[col[e]], 1.0f);
}

// ---------------------------------------------------------------------------
extern "C" void kernel_launch(void* const* d_in, const int* in_sizes, int n_in,
                              void* d_out, int out_size, void* d_ws, size_t ws_size,
                              hipStream_t stream) {
  const float* pos   = (const float*)d_in[0];
  const float* ang   = (const float*)d_in[1];
  const int*   eidx  = (const int*)d_in[2];
  const float* eattr = (const float*)d_in[3];

  const int N = in_sizes[0] / 3;       // 50000
  const int E = in_sizes[3] / ED;      // 1600000
  const int* rowIdx = eidx;
  const int* colIdx = eidx + E;

  // params: pos1@4, pos2@12, ang1@20, ang2@28 ; each {W1,b1,gamma,beta,W2,b2,W3,b3}
  auto P = [&](int base, int k) { return (const float*)d_in[base + k]; };

  // workspace layout (f32)
  float* accP  = (float*)d_ws;                    // [N*64]
  float* accA  = accP + (size_t)N * ED;           // [N*64]
  float* cnt   = accA + (size_t)N * ED;           // [N]
  float* stats = cnt + N;                         // 8*128 (sum/sq x 4 MLPs)
  float* bn    = stats + 8 * HD;                  // 8*128 (s/t x 4 MLPs)
  size_t zeroBytes = ((size_t)2 * N * ED + N + 8 * HD) * sizeof(float);
  hipMemsetAsync(d_ws, 0, zeroBytes, stream);

  const int eBlocks = (E + BM - 1) / BM;
  const int nBlocks = (N + BM - 1) / BM;

  count_edges<<<(E + 255) / 256, 256, 0, stream>>>(colIdx, cnt, E);

  // ---- edge MLPs: stats pass ----
  edge_pass1<3><<<eBlocks, NTHR, 0, stream>>>(pos, eattr, rowIdx,
      P(4, 0), P(4, 1), stats + 0 * HD, stats + 1 * HD, E);
  edge_pass1<4><<<eBlocks, NTHR, 0, stream>>>(ang, eattr, rowIdx,
      P(20, 0), P(20, 1), stats + 2 * HD, stats + 3 * HD, E);
  bn_finalize<<<1, HD, 0, stream>>>(stats + 0 * HD, stats + 1 * HD,
      P(4, 2), P(4, 3), (float)E, bn + 0 * HD, bn + 1 * HD);
  bn_finalize<<<1, HD, 0, stream>>>(stats + 2 * HD, stats + 3 * HD,
      P(20, 2), P(20, 3), (float)E, bn + 2 * HD, bn + 3 * HD);

  // ---- edge MLPs: fused 3-layer + scatter ----
  edge_pass2<3><<<eBlocks, NTHR, 0, stream>>>(pos, eattr, rowIdx, colIdx,
      P(4, 0), P(4, 1), bn + 0 * HD, bn + 1 * HD,
      P(4, 4), P(4, 5), P(4, 6), P(4, 7), accP, E);
  edge_pass2<4><<<eBlocks, NTHR, 0, stream>>>(ang, eattr, rowIdx, colIdx,
      P(20, 0), P(20, 1), bn + 2 * HD, bn + 3 * HD,
      P(20, 4), P(20, 5), P(20, 6), P(20, 7), accA, E);

  // ---- node MLPs: stats pass ----
  node_pass1<3><<<nBlocks, NTHR, 0, stream>>>(pos, accP, cnt,
      P(12, 0), P(12, 1), stats + 4 * HD, stats + 5 * HD, N);
  node_pass1<4><<<nBlocks, NTHR, 0, stream>>>(ang, accA, cnt,
      P(28, 0), P(28, 1), stats + 6 * HD, stats + 7 * HD, N);
  bn_finalize<<<1, HD, 0, stream>>>(stats + 4 * HD, stats + 5 * HD,
      P(12, 2), P(12, 3), (float)N, bn + 4 * HD, bn + 5 * HD);
  bn_finalize<<<1, HD, 0, stream>>>(stats + 6 * HD, stats + 7 * HD,
      P(28, 2), P(28, 3), (float)N, bn + 6 * HD, bn + 7 * HD);

  // ---- node MLPs: fused 3-layer + residual ----
  float* outU   = (float*)d_out;            // [N,3]
  float* outPhi = (float*)d_out + (size_t)N * 3;  // [N,4]
  node_pass2<3><<<nBlocks, NTHR, 0, stream>>>(pos, accP, cnt,
      P(12, 0), P(12, 1), bn + 4 * HD, bn + 5 * HD,
      P(12, 4), P(12, 5), P(12, 6), P(12, 7), outU, N);
  node_pass2<4><<<nBlocks, NTHR, 0, stream>>>(ang, accA, cnt,
      P(28, 0), P(28, 1), bn + 6 * HD, bn + 7 * HD,
      P(28, 4), P(28, 5), P(28, 6), P(28, 7), outPhi, N);
  (void)n_in; (void)out_size; (void)ws_size;
}
